// TweetDisasterClassifierCNN_57002805952719
// MI455X (gfx1250) — matrix operand order repaired and checked by reading, not compiled, not run on hardware
//
#include <hip/hip_runtime.h>
#include <hip/hip_bf16.h>

// ---------------- types ----------------
typedef __bf16 bf16;
typedef __attribute__((ext_vector_type(16))) __bf16 bf16x16;
typedef __attribute__((ext_vector_type(8)))  __bf16 bf16x8;
typedef __attribute__((ext_vector_type(8)))  float  f32x8;

#define B_   64
#define S_   512
#define SP_  520      // padded rows per batch (>= S + k - 1 for k<=5, rounded up)
#define D_   768
#define F_   256
#define NF_  1024     // 4 * F

__device__ __forceinline__ bf16x16 make_frag(const void* p0, const void* p1) {
    union { bf16x16 v; bf16x8 h[2]; } u;
    u.h[0] = *(const bf16x8*)p0;
    u.h[1] = *(const bf16x8*)p1;
    return u.v;
}

#define WMMA_BF16(A, Bf, C) \
    __builtin_amdgcn_wmma_f32_16x16x32_bf16(false, (A), false, (Bf), (short)0, (C), false, false)

// ---------------- kernel 1: stable compaction map + lengths ----------------
// grid = B_, block = 256. inv[b][j] = source position of j-th non-pad token, -1 past L.
__global__ void compact_kernel(const int* __restrict__ x,
                               int* __restrict__ inv, int* __restrict__ Lbuf) {
    __shared__ int sInv[SP_];
    __shared__ int sWave[8];
    const int b = blockIdx.x, tid = threadIdx.x;
    const int wid = tid >> 5, lane = tid & 31;
    for (int j = tid; j < SP_; j += 256) sInv[j] = -1;
    __syncthreads();
    int carry = 0;
    for (int c = 0; c < 2; ++c) {
        const int t = c * 256 + tid;
        const int m = (x[b * S_ + t] != 0) ? 1 : 0;
        const unsigned bal = (unsigned)__ballot(m);
        const int wprefix = __popc(bal & ((1u << lane) - 1u));
        if (lane == 0) sWave[wid] = __popc(bal);
        __syncthreads();
        int wbase = 0, total = 0;
        #pragma unroll
        for (int i = 0; i < 8; ++i) { if (i < wid) wbase += sWave[i]; total += sWave[i]; }
        if (m) sInv[carry + wbase + wprefix] = t;
        carry += total;
        __syncthreads();
    }
    if (tid == 0) Lbuf[b] = carry;
    __syncthreads();
    for (int j = tid; j < SP_; j += 256) inv[b * SP_ + j] = sInv[j];
}

// ---------------- kernel 2: gather h rows -> bf16 hc (zero padded) ----------------
// grid = (SP_, B_), block = 256
__global__ void gather_convert_kernel(const float* __restrict__ h,
                                      const int* __restrict__ inv,
                                      bf16* __restrict__ hcc) {
    const int j = blockIdx.x, b = blockIdx.y, tid = threadIdx.x;
    const int src = inv[b * SP_ + j];
    bf16* dst = hcc + ((size_t)b * SP_ + j) * D_;
    if (src >= 0) {
        const float* s = h + ((size_t)b * S_ + src) * D_;
        for (int d = tid; d < D_; d += 256) dst[d] = (bf16)s[d];
    } else {
        for (int d = tid; d < D_; d += 256) dst[d] = (bf16)0.f;
    }
}

// ---------------- kernel 3: pack conv weights to bf16 WMMA-B layout ----------------
// Packed element index e = ((nTile*nK + ks)*32 + lane)*16 + i
//   lane<16 : column f = nTile*16+lane, K = ks*32 + (i&7) + (i>>3)*16       (i = 0..15)
//   lane>=16: column f = nTile*16+lane-16, K = ks*32 + 8 + (i&7) + (i>>3)*16
// K index kk = j*D + d  maps to  w[f][d][j]  ('OIH' layout: f*D*k + d*k + j)
__global__ void pack_weights_kernel(const float* __restrict__ w2, const float* __restrict__ w3,
                                    const float* __restrict__ w4, const float* __restrict__ w5,
                                    bf16* __restrict__ wpack) {
    const int br = blockIdx.y;
    const int k  = br + 2;
    const int Ktot = D_ * k;
    const size_t nE = (size_t)F_ * Ktot;
    const size_t e = (size_t)blockIdx.x * 256 + threadIdx.x;
    if (e >= nE) return;
    const int i    = (int)(e & 15);
    const int lane = (int)((e >> 4) & 31);
    const size_t rest = e >> 9;                  // nTile*nK + ks
    const int nK = Ktot / 32;
    const int ks = (int)(rest % nK);
    const int nTile = (int)(rest / nK);
    const int f  = nTile * 16 + (lane & 15);
    const int kk = ks * 32 + (lane >> 4) * 8 + (i & 7) + (i >> 3) * 16;
    const int j  = kk / D_;
    const int d  = kk - j * D_;
    const float* w = (br == 0) ? w2 : (br == 1) ? w3 : (br == 2) ? w4 : w5;
    const float v = w[((size_t)f * D_ + d) * k + j];
    const size_t wOff = (size_t)(br * (br + 3) / 2) * D_ * F_;  // prefix of F*Ktot
    wpack[wOff + e] = (bf16)v;
}

// ---------------- kernel 4: conv-as-GEMM (bf16 WMMA) + fused masked max-pool ----------------
// grid = (2 nTiles of 128 filters, 64 batch, 4 branches), block = 256 (8 waves)
// waves: 4 M-groups x 2 N-groups; wave tile = 32 positions x 64 filters
// K-loop is software-pipelined: iteration ks computes on fragments loaded in
// iteration ks-1. __launch_bounds__(256,1) lifts the VGPR cap so the double
// buffer (~190 VGPRs) stays in registers (round-2 build spilled to scratch).
__global__ void __launch_bounds__(256, 1)
conv_gemm_max_kernel(const bf16* __restrict__ hcc,
                     const bf16* __restrict__ wpack,
                     const float* __restrict__ b2, const float* __restrict__ b3,
                     const float* __restrict__ b4, const float* __restrict__ b5,
                     const int* __restrict__ Lbuf,
                     float* __restrict__ feats) {
    const int nTile128 = blockIdx.x;
    const int b        = blockIdx.y;
    const int br       = blockIdx.z;
    const int k  = br + 2;
    const int nK = 24 * k;                        // (D_*k)/32
    const size_t wOff = (size_t)(br * (br + 3) / 2) * D_ * F_;

    const int tid   = threadIdx.x;
    const int lane  = tid & 31;
    const int wave  = tid >> 5;
    const int waveM = wave & 3;
    const int waveN = wave >> 2;
    const int lRow  = lane & 15;
    const int lHalf = lane >> 4;

    const int Lb     = Lbuf[b];
    const int Tvalid = Lb - k + 1;                // conv position t valid iff t < Tvalid (>=4)
    const int nTT    = (Tvalid + 127) / 128;

    const bf16* hcB = hcc + (size_t)b * SP_ * D_;
    const bf16* wp  = wpack + wOff;
    const int nTileBase = nTile128 * 8 + waveN * 4;   // 16-col B tile index, + ni

    // single vector B base + uniform per-ni stride (keeps addressing in 1 VGPR pair)
    const char* bbase0 = (const char*)(wp + (((size_t)nTileBase * nK) * 32 + lane) * 16);
    const int bstride = nK * 1024;                // bytes between consecutive 16-col tiles

    float rmax[4];
    #pragma unroll
    for (int ni = 0; ni < 4; ++ni) rmax[ni] = -__builtin_inff();

    for (int tt = 0; tt < nTT; ++tt) {
        const int t0 = tt * 128 + waveM * 32;
        f32x8 acc[2][4];
        #pragma unroll
        for (int mi = 0; mi < 2; ++mi)
            #pragma unroll
            for (int ni = 0; ni < 4; ++ni)
                acc[mi][ni] = (f32x8)0.f;

        const char* aR0 = (const char*)(hcB + (size_t)(t0 + lRow) * D_);
        const char* aR1 = aR0 + (size_t)16 * D_ * sizeof(bf16);
        const int aoff = lHalf * 16;

        // pipeline prologue: load ks = 0 fragments
        bf16x16 a0c = make_frag(aR0 + aoff, aR0 + aoff + 32);
        bf16x16 a1c = make_frag(aR1 + aoff, aR1 + aoff + 32);
        bf16x16 bc0 = make_frag(bbase0 + 0 * bstride, bbase0 + 0 * bstride + 16);
        bf16x16 bc1 = make_frag(bbase0 + 1 * bstride, bbase0 + 1 * bstride + 16);
        bf16x16 bc2 = make_frag(bbase0 + 2 * bstride, bbase0 + 2 * bstride + 16);
        bf16x16 bc3 = make_frag(bbase0 + 3 * bstride, bbase0 + 3 * bstride + 16);

        for (int ks = 0; ks < nK; ++ks) {
            // issue next iteration's loads first (last iteration over-reads 1 step:
            // A stays inside zero-padded hcc rows; wpack slab is padded by 2KB)
            const int an = (ks + 1) * 64 + aoff;
            const bf16x16 a0n = make_frag(aR0 + an, aR0 + an + 32);
            const bf16x16 a1n = make_frag(aR1 + an, aR1 + an + 32);
            const int bo = (ks + 1) * 1024;
            const bf16x16 bn0 = make_frag(bbase0 + 0 * bstride + bo, bbase0 + 0 * bstride + bo + 16);
            const bf16x16 bn1 = make_frag(bbase0 + 1 * bstride + bo, bbase0 + 1 * bstride + bo + 16);
            const bf16x16 bn2 = make_frag(bbase0 + 2 * bstride + bo, bbase0 + 2 * bstride + bo + 16);
            const bf16x16 bn3 = make_frag(bbase0 + 3 * bstride + bo, bbase0 + 3 * bstride + bo + 16);
            __builtin_prefetch(bbase0 + bo + 1024, 0, 3);

            // compute on current (already-resident) fragments
            acc[0][0] = WMMA_BF16(a0c, bc0, acc[0][0]);
            acc[1][0] = WMMA_BF16(a1c, bc0, acc[1][0]);
            acc[0][1] = WMMA_BF16(a0c, bc1, acc[0][1]);
            acc[1][1] = WMMA_BF16(a1c, bc1, acc[1][1]);
            acc[0][2] = WMMA_BF16(a0c, bc2, acc[0][2]);
            acc[1][2] = WMMA_BF16(a1c, bc2, acc[1][2]);
            acc[0][3] = WMMA_BF16(a0c, bc3, acc[0][3]);
            acc[1][3] = WMMA_BF16(a1c, bc3, acc[1][3]);

            a0c = a0n; a1c = a1n;
            bc0 = bn0; bc1 = bn1; bc2 = bn2; bc3 = bn3;
        }

        // fused masked max over time within this tile
        const int vOff = (lane < 16) ? 0 : 8;     // C layout: lane>=16 holds M = v + 8
        #pragma unroll
        for (int mi = 0; mi < 2; ++mi) {
            const int tBase = t0 + mi * 16 + vOff;
            #pragma unroll
            for (int ni = 0; ni < 4; ++ni) {
                #pragma unroll
                for (int v = 0; v < 8; ++v) {
                    if (tBase + v < Tvalid) rmax[ni] = fmaxf(rmax[ni], acc[mi][ni][v]);
                }
            }
        }
    }

    // merge the two M-halves sharing the same filter column (lane l <-> l+16)
    #pragma unroll
    for (int ni = 0; ni < 4; ++ni)
        rmax[ni] = fmaxf(rmax[ni], __shfl_xor(rmax[ni], 16, 32));

    // reduce across the 4 M-waves via LDS
    __shared__ float sred[2][4][4][16];           // [waveN][waveM][ni][col]
    if (lane < 16) {
        #pragma unroll
        for (int ni = 0; ni < 4; ++ni) sred[waveN][waveM][ni][lane] = rmax[ni];
    }
    __syncthreads();
    if (waveM == 0 && lane < 16) {
        const float* bias = (br == 0) ? b2 : (br == 1) ? b3 : (br == 2) ? b4 : b5;
        #pragma unroll
        for (int ni = 0; ni < 4; ++ni) {
            float m = sred[waveN][0][ni][lane];
            #pragma unroll
            for (int w = 1; w < 4; ++w) m = fmaxf(m, sred[waveN][w][ni][lane]);
            const int f = nTile128 * 128 + waveN * 64 + ni * 16 + lane;
            const float val = fmaxf(m + bias[f], 0.f);     // bias const over t, then ReLU
            feats[(size_t)b * NF_ + br * F_ + f] = val;
        }
    }
}

// ---------------- kernel 5: FC + sigmoid ----------------
// grid = B_, block = 256
__global__ void fc_sigmoid_kernel(const float* __restrict__ feats,
                                  const float* __restrict__ fcw,
                                  const float* __restrict__ fcb,
                                  float* __restrict__ out) {
    __shared__ float s[256];
    const int b = blockIdx.x, tid = threadIdx.x;
    float acc = 0.f;
    for (int i = tid; i < NF_; i += 256) acc += feats[(size_t)b * NF_ + i] * fcw[i];
    s[tid] = acc;
    __syncthreads();
    for (int off = 128; off > 0; off >>= 1) {
        if (tid < off) s[tid] += s[tid + off];
        __syncthreads();
    }
    if (tid == 0) out[b] = 1.f / (1.f + __expf(-(s[0] + fcb[0])));
}

// ---------------- launcher ----------------
extern "C" void kernel_launch(void* const* d_in, const int* in_sizes, int n_in,
                              void* d_out, int out_size, void* d_ws, size_t ws_size,
                              hipStream_t stream) {
    const int*   x    = (const int*)  d_in[0];
    const float* h    = (const float*)d_in[1];
    const float* w2   = (const float*)d_in[2];
    const float* b2   = (const float*)d_in[3];
    const float* w3   = (const float*)d_in[4];
    const float* b3   = (const float*)d_in[5];
    const float* w4   = (const float*)d_in[6];
    const float* b4   = (const float*)d_in[7];
    const float* w5   = (const float*)d_in[8];
    const float* b5   = (const float*)d_in[9];
    const float* fcw  = (const float*)d_in[10];
    const float* fcb  = (const float*)d_in[11];
    float* out = (float*)d_out;

    // workspace layout (256B aligned slabs)
    char* ws = (char*)d_ws;
    size_t off = 0;
    float* feats = (float*)(ws + off); off += (size_t)B_ * NF_ * sizeof(float);     // 256 KB
    int*   Lbuf  = (int*)  (ws + off); off += 256;                                  // 64 ints
    int*   inv   = (int*)  (ws + off); off += (size_t)B_ * SP_ * sizeof(int);       // 130 KB
    bf16*  hcc   = (bf16*) (ws + off); off += (size_t)B_ * SP_ * D_ * sizeof(bf16); // 51 MB
    bf16*  wpack = (bf16*) (ws + off); off += (size_t)14 * D_ * F_ * sizeof(bf16)
                                              + 2048;  // +2KB pad for pipeline over-read
    (void)ws_size; (void)in_sizes; (void)n_in; (void)out_size;

    compact_kernel<<<B_, 256, 0, stream>>>(x, inv, Lbuf);
    gather_convert_kernel<<<dim3(SP_, B_), 256, 0, stream>>>(h, inv, hcc);
    // max elements per branch = F * D * 5 = 983040 -> 3840 blocks of 256
    pack_weights_kernel<<<dim3(3840, 4), 256, 0, stream>>>(w2, w3, w4, w5, wpack);
    conv_gemm_max_kernel<<<dim3(2, B_, 4), 256, 0, stream>>>(hcc, wpack, b2, b3, b4, b5,
                                                             Lbuf, feats);
    fc_sigmoid_kernel<<<B_, 256, 0, stream>>>(feats, fcw, fcb, out);
}
